// MultiHead_21818433863864
// MI455X (gfx1250) — matrix operand order repaired and checked by reading, not compiled, not run on hardware
//
#include <hip/hip_runtime.h>

// ---------------------------------------------------------------------------
// MultiHeadAttention forward for MI455X (gfx1250, wave32, WMMA).
//   1) gemm_bf16_wmma<f32 in, bf16 head-split out> : Q/K/V projections
//      (double-buffered LDS, VGPR prefetch of the next K-slab)
//   2) attn_pass<0> : scores (WMMA bf16) + online softmax rowmax/rowsum
//   3) attn_pass<1> : recompute scores, write normalized attn once (NT store),
//                     ctx = P @ V via WMMA (P bounced through LDS to A-frag)
//      (Q/K tiles staged with GLOBAL_LOAD_ASYNC_TO_LDS_B128, double-buffered)
//   4) gemm_bf16_wmma<bf16 in, f32 out> : output projection + bias
// All matrix math: v_wmma_f32_16x16x32_bf16. Softmax math in f32.
// ---------------------------------------------------------------------------

typedef __attribute__((ext_vector_type(16))) __bf16 v16bf;
typedef __attribute__((ext_vector_type(8)))  float  v8f;
typedef __attribute__((ext_vector_type(4)))  int    v4i_ty;

union FragB16 { v16bf v; unsigned int u[8]; };

__device__ __forceinline__ unsigned short f32_to_bf16_rne(float f) {
  unsigned int u = __builtin_bit_cast(unsigned int, f);
  u += 0x7FFFu + ((u >> 16) & 1u);
  return (unsigned short)(u >> 16);
}

#define SEQ    2048
#define DM     1024
#define NHEAD  16
#define DEPTH  64

// ---- CDNA5 async global->LDS copy (ASYNCcnt path), with safe fallback ----
#if __has_builtin(__builtin_amdgcn_global_load_async_to_lds_b128)
#define HAVE_ASYNC_LDS 1
#else
#define HAVE_ASYNC_LDS 0
#endif
#if __has_builtin(__builtin_amdgcn_s_wait_asynccnt)
#define HAVE_WAIT_ASYNC 1
#else
#define HAVE_WAIT_ASYNC 0
#endif

__device__ __forceinline__ void async_copy_b128(void* lds_dst, const void* g_src) {
#if HAVE_ASYNC_LDS
  __builtin_amdgcn_global_load_async_to_lds_b128(
      (v4i_ty*)g_src, (v4i_ty*)lds_dst, 0, 0);
#else
  *(uint4*)lds_dst = *(const uint4*)g_src;
#endif
}

__device__ __forceinline__ void wait_async_all() {
#if HAVE_ASYNC_LDS
#if HAVE_WAIT_ASYNC
  __builtin_amdgcn_s_wait_asynccnt(0);
#else
  asm volatile("s_wait_asynccnt 0x0" ::: "memory");
#endif
#endif
}

// ---------------------------------------------------------------------------
// GEMM: Y[M=4096,1024] = X @ W + bias. Double-buffered LDS (2x16KB).
// Block 256 threads = 8 waves (2x4); block tile 128x128; wave tile 64x32.
// ---------------------------------------------------------------------------
template <bool XBF16, bool OUTF32>
__global__ __launch_bounds__(256) void gemm_bf16_wmma(
    const void* __restrict__ Xraw,      // f32 or bf16 [4096,1024]
    const float* __restrict__ W,        // f32 [1024,1024]
    const float* __restrict__ bias,     // f32 [1024]
    unsigned short* __restrict__ Ybf,   // bf16 head-split (mode 0)
    float* __restrict__ Yf)             // f32 (mode 1)
{
  __shared__ __attribute__((aligned(16))) unsigned short ldsA[2][128 * 32];
  __shared__ __attribute__((aligned(16))) unsigned short ldsBt[2][128 * 32];

  const int tid    = threadIdx.x;
  const int lane   = tid & 31;
  const int wave   = tid >> 5;
  const int halfId = lane >> 4;
  const int l16    = lane & 15;
  const int warpM  = wave >> 2;  // 0..1
  const int warpN  = wave & 3;   // 0..3
  const int nBase  = blockIdx.x * 128;
  const int mBase  = blockIdx.y * 128;

  v8f acc[4][2];
#pragma unroll
  for (int i = 0; i < 4; ++i)
#pragma unroll
    for (int j = 0; j < 2; ++j) acc[i][j] = (v8f)0.0f;

  float4       xa[4];   // f32 A prefetch
  unsigned int xu[8];   // bf16 A prefetch
  float4       wb[4];   // W prefetch

  auto loadRegs = [&](int kk0) {
    if constexpr (XBF16) {
      const unsigned int* Xu = (const unsigned int*)Xraw;  // 2 bf16/dword
#pragma unroll
      for (int p = 0; p < 8; ++p) {
        int linear = p * 256 + tid;            // 2048 dwords
        int row = linear >> 4, c = linear & 15;
        xu[p] = Xu[(size_t)(mBase + row) * (DM / 2) + (kk0 >> 1) + c];
      }
    } else {
      const float* Xf = (const float*)Xraw;
#pragma unroll
      for (int p = 0; p < 4; ++p) {
        int linear = p * 256 + tid;            // 1024 float4
        int row = linear >> 3, c4 = (linear & 7) * 4;
        xa[p] = *(const float4*)&Xf[(size_t)(mBase + row) * DM + kk0 + c4];
      }
    }
#pragma unroll
    for (int p = 0; p < 4; ++p) {
      int linear = p * 256 + tid;              // 1024 float4
      int k = linear >> 5, n4 = (linear & 31) * 4;
      wb[p] = *(const float4*)&W[(size_t)(kk0 + k) * DM + nBase + n4];
    }
  };

  auto storeLds = [&](int bi) {
    if constexpr (XBF16) {
#pragma unroll
      for (int p = 0; p < 8; ++p) {
        int linear = p * 256 + tid;
        int row = linear >> 4, c = linear & 15;
        ((unsigned int*)ldsA[bi])[row * 16 + c] = xu[p];
      }
    } else {
#pragma unroll
      for (int p = 0; p < 4; ++p) {
        int linear = p * 256 + tid;
        int row = linear >> 3, c4 = (linear & 7) * 4;
        unsigned int lo = ((unsigned int)f32_to_bf16_rne(xa[p].y) << 16) | f32_to_bf16_rne(xa[p].x);
        unsigned int hi = ((unsigned int)f32_to_bf16_rne(xa[p].w) << 16) | f32_to_bf16_rne(xa[p].z);
        unsigned int* dst = (unsigned int*)&ldsA[bi][row * 32 + c4];
        dst[0] = lo; dst[1] = hi;
      }
    }
#pragma unroll
    for (int p = 0; p < 4; ++p) {
      int linear = p * 256 + tid;
      int k = linear >> 5, n4 = (linear & 31) * 4;
      ldsBt[bi][(n4 + 0) * 32 + k] = f32_to_bf16_rne(wb[p].x);
      ldsBt[bi][(n4 + 1) * 32 + k] = f32_to_bf16_rne(wb[p].y);
      ldsBt[bi][(n4 + 2) * 32 + k] = f32_to_bf16_rne(wb[p].z);
      ldsBt[bi][(n4 + 3) * 32 + k] = f32_to_bf16_rne(wb[p].w);
    }
  };

  loadRegs(0);
  storeLds(0);

  for (int t = 0; t < DM / 32; ++t) {
    const int bi = t & 1;
    __syncthreads();
    if (t + 1 < DM / 32) loadRegs((t + 1) * 32);  // prefetch next slab to VGPRs

    FragB16 afr[4];
#pragma unroll
    for (int fm = 0; fm < 4; ++fm) {
      int m = warpM * 64 + fm * 16 + l16;
#pragma unroll
      for (int j = 0; j < 8; ++j) {
        int kk = 2 * j + halfId * 8 + ((j >= 4) ? 8 : 0);  // A layout (ISA 7.12.2)
        afr[fm].u[j] = *(const unsigned int*)&ldsA[bi][m * 32 + kk];
      }
    }
    FragB16 bfr[2];
#pragma unroll
    for (int fn = 0; fn < 2; ++fn) {
      int n = warpN * 32 + fn * 16 + l16;
#pragma unroll
      for (int j = 0; j < 8; ++j)
        bfr[fn].u[j] = *(const unsigned int*)&ldsBt[bi][n * 32 + halfId * 8 + 2 * j + ((j >= 4) ? 8 : 0)];
    }
#pragma unroll
    for (int fm = 0; fm < 4; ++fm)
#pragma unroll
      for (int fn = 0; fn < 2; ++fn)
        acc[fm][fn] = __builtin_amdgcn_wmma_f32_16x16x32_bf16(
            false, afr[fm].v, false, bfr[fn].v, (short)0, acc[fm][fn], false, false);

    if (t + 1 < DM / 32) storeLds(bi ^ 1);        // fill other buffer
  }

  // ---- epilogue ----
#pragma unroll
  for (int fn = 0; fn < 2; ++fn) {
    int col = nBase + warpN * 32 + fn * 16 + l16;
    float bv = bias[col];
#pragma unroll
    for (int fm = 0; fm < 4; ++fm) {
#pragma unroll
      for (int j = 0; j < 8; ++j) {
        int row = mBase + warpM * 64 + fm * 16 + j + 8 * halfId;
        float v = acc[fm][fn][j] + bv;
        if (OUTF32) {
          __builtin_nontemporal_store(v, &Yf[(size_t)row * DM + col]);
        } else {
          int bb = row >> 11, s = row & (SEQ - 1);
          Ybf[(((size_t)(bb * NHEAD + (col >> 6)) * SEQ) + s) * DEPTH + (col & 63)] =
              f32_to_bf16_rne(v);
        }
      }
    }
  }
}

// ---------------------------------------------------------------------------
// Attention. One block = one (b, h, 64-row q tile); 128 threads = 4 waves;
// each wave owns 16 q rows. K/V tiles double-buffered; Q/K staged async.
// PASS 0: online rowmax/rowsum.  PASS 1: write attn (NT) + accumulate ctx.
// ---------------------------------------------------------------------------
template <int PASS>
__global__ __launch_bounds__(128) void attn_pass(
    const unsigned short* __restrict__ qh,   // bf16 [B,H,S,DEPTH]
    const unsigned short* __restrict__ kh,   // bf16 [B,H,S,DEPTH]
    const unsigned short* __restrict__ vh,   // bf16 [B,H,S,DEPTH]
    const float* __restrict__ mask,          // f32 [B,S]
    float* __restrict__ rowmax,              // f32 [B*H*S]
    float* __restrict__ rowsum,              // f32 [B*H*S]
    float* __restrict__ attn,                // f32 [B,H,S,S]
    unsigned short* __restrict__ ctxb)       // bf16 [B*S, DM] head-merged
{
  __shared__ __attribute__((aligned(16))) unsigned short lq[64 * 64];      // 8KB
  __shared__ __attribute__((aligned(16))) unsigned short lk[2][64 * 64];   // 16KB
  __shared__ __attribute__((aligned(16))) unsigned short lvt[2][64 * 64];  // 16KB
  __shared__ __attribute__((aligned(16))) unsigned short lp[4][16 * 64];   // 8KB

  const int tid    = threadIdx.x;
  const int lane   = tid & 31;
  const int wave   = tid >> 5;
  const int halfId = lane >> 4;
  const int l16    = lane & 15;
  const int qt = blockIdx.x, h = blockIdx.y, b = blockIdx.z;
  const int bh = b * NHEAD + h;
  const int qbase = qt * 64;

  auto stageK = [&](int buf, int kt) {  // async 8KB, 4 x b128 per thread
    const char* ksrc = (const char*)(kh + ((size_t)bh * SEQ + kt * 64) * DEPTH);
#pragma unroll
    for (int i = 0; i < 4; ++i) {
      int lin = i * 128 + tid;
      async_copy_b128((char*)lk[buf] + lin * 16, ksrc + lin * 16);
    }
  };
  auto stageV = [&](int buf, int kt) {  // manual transposed store [d][key]
    const unsigned short* vsrc = vh + ((size_t)bh * SEQ + kt * 64) * DEPTH;
#pragma unroll
    for (int i = 0; i < 32; ++i) {
      int lin = i * 128 + tid;
      int key = lin >> 6, d = lin & 63;
      lvt[buf][d * 64 + key] = vsrc[lin];
    }
  };

  // prologue: async q tile + first k tile, manual first v tile
  {
    const char* qsrc = (const char*)(qh + ((size_t)bh * SEQ + qbase) * DEPTH);
#pragma unroll
    for (int i = 0; i < 4; ++i) {
      int lin = i * 128 + tid;
      async_copy_b128((char*)lq + lin * 16, qsrc + lin * 16);
    }
  }
  stageK(0, 0);
  stageV(0, 0);

  float rm[8], rl[8], smax[8], sinv[8];
  if (PASS == 0) {
#pragma unroll
    for (int j = 0; j < 8; ++j) { rm[j] = -1e30f; rl[j] = 0.0f; }
  } else {
#pragma unroll
    for (int j = 0; j < 8; ++j) {
      int row = qbase + wave * 16 + j + 8 * halfId;
      smax[j] = rowmax[(size_t)bh * SEQ + row];
      sinv[j] = 1.0f / rowsum[(size_t)bh * SEQ + row];
    }
  }

  v8f cacc[4];
#pragma unroll
  for (int i = 0; i < 4; ++i) cacc[i] = (v8f)0.0f;

  for (int kt = 0; kt < SEQ / 64; ++kt) {
    const int bi = kt & 1;
    const int keybase = kt * 64;
    wait_async_all();
    __syncthreads();
    if (kt + 1 < SEQ / 64) {   // issue next tile while computing this one
      stageK(bi ^ 1, kt + 1);
      stageV(bi ^ 1, kt + 1);
    }

    // ---- scores: S(16x64) = Q(16x64) @ K^T, K-dim = depth (2 steps of 32) ----
    v8f sc[4];
#pragma unroll
    for (int i = 0; i < 4; ++i) sc[i] = (v8f)0.0f;
#pragma unroll
    for (int ks = 0; ks < 2; ++ks) {
      FragB16 a;
      int m = wave * 16 + l16;
#pragma unroll
      for (int j = 0; j < 8; ++j) {
        int kk = ks * 32 + 2 * j + halfId * 8 + ((j >= 4) ? 8 : 0);
        a.u[j] = *(const unsigned int*)&lq[m * 64 + kk];
      }
#pragma unroll
      for (int fn = 0; fn < 4; ++fn) {
        FragB16 bb;
        int key = fn * 16 + l16;
#pragma unroll
        for (int j = 0; j < 8; ++j)
          bb.u[j] = *(const unsigned int*)&lk[bi][key * 64 + ks * 32 + halfId * 8 + 2 * j + ((j >= 4) ? 8 : 0)];
        sc[fn] = __builtin_amdgcn_wmma_f32_16x16x32_bf16(
            false, a.v, false, bb.v, (short)0, sc[fn], false, false);
      }
    }
    // scale + mask
    float mk[4];
#pragma unroll
    for (int fn = 0; fn < 4; ++fn)
      mk[fn] = mask[b * SEQ + keybase + fn * 16 + l16] * -1e9f;
#pragma unroll
    for (int fn = 0; fn < 4; ++fn)
#pragma unroll
      for (int j = 0; j < 8; ++j)
        sc[fn][j] = sc[fn][j] * 0.125f + mk[fn];

    if (PASS == 0) {
#pragma unroll
      for (int j = 0; j < 8; ++j) {
        float tmax = fmaxf(fmaxf(sc[0][j], sc[1][j]), fmaxf(sc[2][j], sc[3][j]));
#pragma unroll
        for (int off = 1; off < 16; off <<= 1)
          tmax = fmaxf(tmax, __shfl_xor(tmax, off, 32));
        float mnew = fmaxf(rm[j], tmax);
        float corr = __expf(rm[j] - mnew);
        float ls = __expf(sc[0][j] - mnew) + __expf(sc[1][j] - mnew) +
                   __expf(sc[2][j] - mnew) + __expf(sc[3][j] - mnew);
#pragma unroll
        for (int off = 1; off < 16; off <<= 1)
          ls += __shfl_xor(ls, off, 32);
        rl[j] = rl[j] * corr + ls;
        rm[j] = mnew;
      }
    } else {
      // normalized probabilities -> attn (streamed, NT) + LDS bounce to A-frag
#pragma unroll
      for (int fn = 0; fn < 4; ++fn) {
#pragma unroll
        for (int j = 0; j < 8; ++j) {
          float p = __expf(sc[fn][j] - smax[j]) * sinv[j];
          int row = qbase + wave * 16 + j + 8 * halfId;
          __builtin_nontemporal_store(
              p, &attn[((size_t)bh * SEQ + row) * SEQ + keybase + fn * 16 + l16]);
          lp[wave][(j + 8 * halfId) * 64 + fn * 16 + l16] = f32_to_bf16_rne(p);
        }
      }
      // ctx += P(16 x 64keys) @ V(64keys x 64depth), 2 K-steps of 32
#pragma unroll
      for (int ks2 = 0; ks2 < 2; ++ks2) {
        FragB16 a;
#pragma unroll
        for (int j = 0; j < 8; ++j) {
          int kk = ks2 * 32 + 2 * j + halfId * 8 + ((j >= 4) ? 8 : 0);
          a.u[j] = *(const unsigned int*)&lp[wave][l16 * 64 + kk];
        }
#pragma unroll
        for (int fn2 = 0; fn2 < 4; ++fn2) {
          FragB16 bb;
          int d = fn2 * 16 + l16;
#pragma unroll
          for (int j = 0; j < 8; ++j)
            bb.u[j] = *(const unsigned int*)&lvt[bi][d * 64 + ks2 * 32 + halfId * 8 + 2 * j + ((j >= 4) ? 8 : 0)];
          cacc[fn2] = __builtin_amdgcn_wmma_f32_16x16x32_bf16(
              false, a.v, false, bb.v, (short)0, cacc[fn2], false, false);
        }
      }
    }
    __syncthreads();
  }

  if (PASS == 0) {
    if (l16 == 0) {
#pragma unroll
      for (int j = 0; j < 8; ++j) {
        int row = qbase + wave * 16 + j + 8 * halfId;
        rowmax[(size_t)bh * SEQ + row] = rm[j];
        rowsum[(size_t)bh * SEQ + row] = rl[j];
      }
    }
  } else {
#pragma unroll
    for (int fn2 = 0; fn2 < 4; ++fn2) {
#pragma unroll
      for (int j = 0; j < 8; ++j) {
        int row = qbase + wave * 16 + j + 8 * halfId;
        int d = fn2 * 16 + l16;
        ctxb[((size_t)(b * SEQ + row)) * DM + h * DEPTH + d] = f32_to_bf16_rne(cacc[fn2][j]);
      }
    }
  }
}

// ---------------------------------------------------------------------------
extern "C" void kernel_launch(void* const* d_in, const int* in_sizes, int n_in,
                              void* d_out, int out_size, void* d_ws, size_t ws_size,
                              hipStream_t stream) {
  const float* q    = (const float*)d_in[0];
  const float* k    = (const float*)d_in[1];
  const float* v    = (const float*)d_in[2];
  const float* mask = (const float*)d_in[3];
  const float* Wq   = (const float*)d_in[4];
  const float* bq   = (const float*)d_in[5];
  const float* Wk   = (const float*)d_in[6];
  const float* bk   = (const float*)d_in[7];
  const float* Wv   = (const float*)d_in[8];
  const float* bv   = (const float*)d_in[9];
  const float* Wo   = (const float*)d_in[10];
  const float* bo   = (const float*)d_in[11];

  float* out  = (float*)d_out;                       // [2,2048,1024]
  float* attn = out + (size_t)2 * SEQ * DM;          // [2,16,2048,2048]

  const size_t MROWS = (size_t)2 * SEQ;              // 4096
  unsigned short* qh  = (unsigned short*)d_ws;       // bf16 [B,H,S,DEPTH]
  unsigned short* kh  = qh + MROWS * DM;
  unsigned short* vh  = kh + MROWS * DM;
  unsigned short* ctx = vh + MROWS * DM;             // bf16 [B*S, DM]
  float* rowmax = (float*)(ctx + MROWS * DM);        // [B*H*S]
  float* rowsum = rowmax + (size_t)2 * NHEAD * SEQ;

  dim3 gGemm(DM / 128, MROWS / 128);                 // (8, 32)
  gemm_bf16_wmma<false, false><<<gGemm, 256, 0, stream>>>(q, Wq, bq, qh, nullptr);
  gemm_bf16_wmma<false, false><<<gGemm, 256, 0, stream>>>(k, Wk, bk, kh, nullptr);
  gemm_bf16_wmma<false, false><<<gGemm, 256, 0, stream>>>(v, Wv, bv, vh, nullptr);

  dim3 gAttn(SEQ / 64, NHEAD, 2);                    // (32, 16, 2)
  attn_pass<0><<<gAttn, 128, 0, stream>>>(qh, kh, vh, mask, rowmax, rowsum, attn, ctx);
  attn_pass<1><<<gAttn, 128, 0, stream>>>(qh, kh, vh, mask, rowmax, rowsum, attn, ctx);

  gemm_bf16_wmma<true, true><<<gGemm, 256, 0, stream>>>(ctx, Wo, bo, nullptr, out);
}
